// PAM_ModuleM_3289944949428
// MI455X (gfx1250) — matrix operand order repaired and checked by reading, not compiled, not run on hardware
//
#include <hip/hip_runtime.h>

// ---------------------------------------------------------------------------
// PAM attention module on gfx1250 (MI455X), wave32 + WMMA f32_16x16x32_f16.
//   B=8, N=2048, C=320, CQ=64
// Pipeline (workspace holds f16 intermediates):
//   K1: q,k = x*Wq^T,x*Wk^T (N,64 row-major);  vT = (x*Wv^T)^T  (C,N)
//   K2: E = q*k^T per 16-row strip (held in 128KB LDS), row softmax,
//       write normalized weights TRANSPOSED: Wt[n][k]  (f16)
//   K3: Out = Wt (N,N) x V  (K=2048), 2x2 register tiling, y = gamma*Out + x
// All operand fragments load as 2x b128 per lane (contiguous K groups).
// ---------------------------------------------------------------------------

#define BATCH 8
#define NPOS  2048
#define CH    320
#define CQ    64

typedef __attribute__((ext_vector_type(16))) _Float16 v16h;
typedef __attribute__((ext_vector_type(8)))  float    v8f;

union FragU { v16h v; unsigned int u[8]; uint4 q[2]; };

static __device__ __forceinline__ unsigned short f2h(float f) {
  union { _Float16 h; unsigned short u; } c; c.h = (_Float16)f; return c.u;
}
static __device__ __forceinline__ unsigned int pk2h(float a, float b) {
  return (unsigned int)f2h(a) | ((unsigned int)f2h(b) << 16);
}

static __device__ __forceinline__ v8f wmma_acc(v16h a, v16h b, v8f c) {
  // D = A(16x32 f16) * B(32x16 f16) + C(16x16 f32)
  return __builtin_amdgcn_wmma_f32_16x16x32_f16(
      /*neg_a=*/false, a, /*neg_b=*/false, b,
      /*c_mod=*/(short)0, c, /*reuse_a=*/false, /*reuse_b=*/false);
}

// A fragment (16x32 f16) from row-major source (row stride ld, K contiguous).
// ISA layout: lane<16 row r: V0..3 K=0..7, V4..7 K=16..23;
//             lane>=16     : V0..3 K=8..15, V4..7 K=24..31.
// => exactly two 16B loads per lane at k = half*8 and 16+half*8.
static __device__ __forceinline__ v16h load_a16(const unsigned short* base,
                                                int ld, int lane) {
  FragU f;
  const int half = lane >> 4;
  const unsigned short* p = base + (size_t)(lane & 15) * ld + half * 8;
  f.q[0] = *(const uint4*)(p);
  f.q[1] = *(const uint4*)(p + 16);
  return f.v;
}

// B fragment (32x16 f16) from B^T row-major (rows = N columns, K contiguous).
// ISA layout: lane<16 col n: V0..3 K=0..7, V4..7 K=16..23; lane>=16: K=8..15, 24..31.
// => two 16B loads per lane at k = half*8 and 16+half*8.
static __device__ __forceinline__ v16h load_b16(const unsigned short* base,
                                                int ld, int lane) {
  FragU f;
  const unsigned short* p =
      base + (size_t)(lane & 15) * ld + ((lane >> 4) << 3);
  f.q[0] = *(const uint4*)(p);
  f.q[1] = *(const uint4*)(p + 16);
  return f.v;
}

// Same fragments, converting from f32 sources via float4 + v_cvt_pk_f16_f32.
static __device__ __forceinline__ v16h load_a32cvt(const float* base, int ld,
                                                   int lane) {
  FragU f;
  const int half = lane >> 4;
  const float* p = base + (size_t)(lane & 15) * ld + half * 8;
  float4 g0 = *(const float4*)(p);
  float4 g1 = *(const float4*)(p + 4);
  float4 g2 = *(const float4*)(p + 16);
  float4 g3 = *(const float4*)(p + 20);
  f.u[0] = pk2h(g0.x, g0.y); f.u[1] = pk2h(g0.z, g0.w);
  f.u[2] = pk2h(g1.x, g1.y); f.u[3] = pk2h(g1.z, g1.w);
  f.u[4] = pk2h(g2.x, g2.y); f.u[5] = pk2h(g2.z, g2.w);
  f.u[6] = pk2h(g3.x, g3.y); f.u[7] = pk2h(g3.z, g3.w);
  return f.v;
}

static __device__ __forceinline__ v16h load_b32cvt(const float* base, int ld,
                                                   int lane) {
  FragU f;
  const float* p = base + (size_t)(lane & 15) * ld + ((lane >> 4) << 3);
  float4 g0 = *(const float4*)(p);
  float4 g1 = *(const float4*)(p + 4);
  float4 g2 = *(const float4*)(p + 16);
  float4 g3 = *(const float4*)(p + 20);
  f.u[0] = pk2h(g0.x, g0.y); f.u[1] = pk2h(g0.z, g0.w);
  f.u[2] = pk2h(g1.x, g1.y); f.u[3] = pk2h(g1.z, g1.w);
  f.u[4] = pk2h(g2.x, g2.y); f.u[5] = pk2h(g2.z, g2.w);
  f.u[6] = pk2h(g3.x, g3.y); f.u[7] = pk2h(g3.z, g3.w);
  return f.v;
}

// ---------------------------------------------------------------------------
// K1: fused QKV projection.  grid=(B, N/16, 7), block=128 (4 waves).
// Wave handles output-channel tile ot = z*4+wave (28 tiles of 16 over 448).
// ---------------------------------------------------------------------------
__global__ void k_qkv_proj(const float* __restrict__ x,
                           const float* __restrict__ Wq,
                           const float* __restrict__ bq,
                           const float* __restrict__ Wk,
                           const float* __restrict__ bk,
                           const float* __restrict__ Wv,
                           const float* __restrict__ bv,
                           unsigned short* __restrict__ q,
                           unsigned short* __restrict__ kp,
                           unsigned short* __restrict__ vT) {
  const int b    = blockIdx.x;
  const int n0   = blockIdx.y * 16;
  const int wave = threadIdx.x >> 5;
  const int lane = threadIdx.x & 31;
  const int o0   = (blockIdx.z * 4 + wave) * 16;  // 0..447

  const float* W;  const float* bias;  int orow;
  if (o0 < 64)        { W = Wq; bias = bq; orow = o0; }
  else if (o0 < 128)  { W = Wk; bias = bk; orow = o0 - 64; }
  else                { W = Wv; bias = bv; orow = o0 - 128; }

  const float* xA = x + ((size_t)b * NPOS + n0) * CH;

  v8f acc = {0.f, 0.f, 0.f, 0.f, 0.f, 0.f, 0.f, 0.f};
  for (int kk = 0; kk < CH; kk += 32) {
    v16h a  = load_a32cvt(xA + kk, CH, lane);
    v16h bf = load_b32cvt(W + (size_t)orow * CH + kk, CH, lane);
    acc = wmma_acc(a, bf, acc);
  }

  const int col = lane & 15, half = lane >> 4;
  const float bias_v = bias[orow + col];
#pragma unroll
  for (int r = 0; r < 8; ++r) acc[r] += bias_v;

  if (o0 < 128) {
    // q / k: row-major (N, 64), K-contiguous for later A/B^T use.
    unsigned short* dst = (o0 < 64) ? q : kp;
#pragma unroll
    for (int r = 0; r < 8; ++r) {
      const int n = n0 + half * 8 + r;
      dst[((size_t)b * NPOS + n) * CQ + orow + col] = f2h(acc[r]);
    }
  } else {
    // vT: [b][c][n] — C/D layout gives 8 consecutive n per lane: one b128.
    union { unsigned short s[8]; uint4 v; } pk;
#pragma unroll
    for (int r = 0; r < 8; ++r) pk.s[r] = f2h(acc[r]);
    const size_t off =
        ((size_t)b * CH + orow + col) * NPOS + n0 + half * 8;
    *(uint4*)(vT + off) = pk.v;
  }
}

// ---------------------------------------------------------------------------
// K2: energy strip + row softmax + transposed weight store.
// grid=(B, N/16), block=256 (8 waves).  Dynamic LDS: 16x2048 f32 strip
// (legal on CDNA5: 320KB LDS per WGP).
// ---------------------------------------------------------------------------
__global__ void k_energy_softmax(const unsigned short* __restrict__ q,
                                 const unsigned short* __restrict__ kp,
                                 unsigned short* __restrict__ Wt) {
  extern __shared__ float smem[];
  float* strip  = smem;                       // [n][k] : 2048*16
  float* red    = smem + NPOS * 16;           // 16x16 partials
  float* rowmax = red + 256;                  // [16]
  float* rowsc  = rowmax + 16;                // [16] = 1/sum

  const int b    = blockIdx.x;
  const int kr0  = blockIdx.y * 16;           // E row block (source positions)
  const int wave = threadIdx.x >> 5;
  const int lane = threadIdx.x & 31;
  const int col  = lane & 15, half = lane >> 4;

  const unsigned short* qbase = q + ((size_t)b * NPOS + kr0) * CQ;
  const v16h a0 = load_a16(qbase + 0,  CQ, lane);
  const v16h a1 = load_a16(qbase + 32, CQ, lane);

  for (int nt = wave; nt < NPOS / 16; nt += 8) {
    const unsigned short* kb = kp + ((size_t)b * NPOS + nt * 16) * CQ;
    v16h b0 = load_b16(kb + 0,  CQ, lane);
    v16h b1 = load_b16(kb + 32, CQ, lane);
    v8f acc = {0.f, 0.f, 0.f, 0.f, 0.f, 0.f, 0.f, 0.f};
    acc = wmma_acc(a0, b0, acc);
    acc = wmma_acc(a1, b1, acc);
    // store strip[n][k]: lane owns column n, rows k = half*8 + 0..7
    const int base = (nt * 16 + col) * 16 + half * 8;
    *(float4*)&strip[base]     = make_float4(acc[0], acc[1], acc[2], acc[3]);
    *(float4*)&strip[base + 4] = make_float4(acc[4], acc[5], acc[6], acc[7]);
  }
  __syncthreads();

  // Row max over n (row = E row k, i.e. strip index [n][k]).
  const int kl = threadIdx.x & 15, ch = threadIdx.x >> 4;
  float m = -3.4e38f;
  for (int n = ch; n < NPOS; n += 16) m = fmaxf(m, strip[n * 16 + kl]);
  red[kl * 16 + ch] = m;
  __syncthreads();
  if (threadIdx.x < 16) {
    float mm = red[threadIdx.x * 16];
#pragma unroll
    for (int i = 1; i < 16; ++i) mm = fmaxf(mm, red[threadIdx.x * 16 + i]);
    rowmax[threadIdx.x] = mm;
  }
  __syncthreads();

  // Row sum of exp.
  const float rm = rowmax[kl];
  float s = 0.f;
  for (int n = ch; n < NPOS; n += 16) s += __expf(strip[n * 16 + kl] - rm);
  red[kl * 16 + ch] = s;
  __syncthreads();
  if (threadIdx.x < 16) {
    float ss = 0.f;
#pragma unroll
    for (int i = 0; i < 16; ++i) ss += red[threadIdx.x * 16 + i];
    rowsc[threadIdx.x] = 1.f / ss;
  }
  __syncthreads();

  // Normalized exp, written transposed: Wt[b][n][kr0+k], f16, 2x b128/row.
  float rmx[16], rsc[16];
#pragma unroll
  for (int i = 0; i < 16; ++i) { rmx[i] = rowmax[i]; rsc[i] = rowsc[i]; }
  for (int n = threadIdx.x; n < NPOS; n += 256) {
    unsigned int uu[8];
#pragma unroll
    for (int j = 0; j < 8; ++j) {
      const float w0 = __expf(strip[n * 16 + 2 * j]     - rmx[2 * j])     * rsc[2 * j];
      const float w1 = __expf(strip[n * 16 + 2 * j + 1] - rmx[2 * j + 1]) * rsc[2 * j + 1];
      uu[j] = pk2h(w0, w1);
    }
    uint4* dst = (uint4*)(Wt + ((size_t)b * NPOS + n) * NPOS + kr0);
    dst[0] = make_uint4(uu[0], uu[1], uu[2], uu[3]);
    dst[1] = make_uint4(uu[4], uu[5], uu[6], uu[7]);
  }
}

// ---------------------------------------------------------------------------
// K3: Out = Wt (N,N) x V, K=2048; y = gamma*Out + x.
// grid=(B, N/32, C/32), block=32 (1 wave); 2x2 register tiling per wave:
// 4 WMMA per 8 b128 loads, A/B fragments reused across the 2x2 tile.
// ---------------------------------------------------------------------------
__global__ void k_attn_out(const unsigned short* __restrict__ Wt,
                           const unsigned short* __restrict__ vT,
                           const float* __restrict__ x,
                           const float* __restrict__ gamma,
                           float* __restrict__ y) {
  const int b    = blockIdx.x;
  const int n0   = blockIdx.y * 32;
  const int c0   = blockIdx.z * 32;
  const int lane = threadIdx.x & 31;

  const unsigned short* A0 = Wt + ((size_t)b * NPOS + n0) * NPOS;
  const unsigned short* A1 = A0 + (size_t)16 * NPOS;
  const unsigned short* B0 = vT + ((size_t)b * CH + c0) * NPOS;
  const unsigned short* B1 = B0 + (size_t)16 * NPOS;

  v8f acc00 = {0.f, 0.f, 0.f, 0.f, 0.f, 0.f, 0.f, 0.f};
  v8f acc01 = acc00, acc10 = acc00, acc11 = acc00;

  for (int kk = 0; kk < NPOS; kk += 32) {
    v16h a0 = load_a16(A0 + kk, NPOS, lane);
    v16h a1 = load_a16(A1 + kk, NPOS, lane);
    v16h b0 = load_b16(B0 + kk, NPOS, lane);
    v16h b1 = load_b16(B1 + kk, NPOS, lane);
    acc00 = wmma_acc(a0, b0, acc00);
    acc01 = wmma_acc(a0, b1, acc01);
    acc10 = wmma_acc(a1, b0, acc10);
    acc11 = wmma_acc(a1, b1, acc11);
  }

  const float g = gamma[0];
  const int col = lane & 15, half = lane >> 4;
#pragma unroll
  for (int r = 0; r < 8; ++r) {
    const int n  = n0 + half * 8 + r;
    const int nn = n + 16;
    const size_t i00 = ((size_t)b * NPOS + n)  * CH + c0 + col;
    const size_t i01 = i00 + 16;
    const size_t i10 = ((size_t)b * NPOS + nn) * CH + c0 + col;
    const size_t i11 = i10 + 16;
    y[i00] = g * acc00[r] + x[i00];
    y[i01] = g * acc01[r] + x[i01];
    y[i10] = g * acc10[r] + x[i10];
    y[i11] = g * acc11[r] + x[i11];
  }
}

// ---------------------------------------------------------------------------
extern "C" void kernel_launch(void* const* d_in, const int* in_sizes, int n_in,
                              void* d_out, int out_size, void* d_ws,
                              size_t ws_size, hipStream_t stream) {
  (void)in_sizes; (void)n_in; (void)out_size; (void)ws_size;
  const float* x     = (const float*)d_in[0];
  const float* Wq    = (const float*)d_in[1];
  const float* bq    = (const float*)d_in[2];
  const float* Wk    = (const float*)d_in[3];
  const float* bk    = (const float*)d_in[4];
  const float* Wv    = (const float*)d_in[5];
  const float* bv    = (const float*)d_in[6];
  const float* gamma = (const float*)d_in[7];
  float* y = (float*)d_out;

  // Workspace layout (f16 as u16): q, k (B,N,64); vT (B,C,N); Wt (B,N,N).
  unsigned short* q  = (unsigned short*)d_ws;
  unsigned short* kp = q  + (size_t)BATCH * NPOS * CQ;
  unsigned short* vT = kp + (size_t)BATCH * NPOS * CQ;
  unsigned short* Wt = vT + (size_t)BATCH * CH * NPOS;

  k_qkv_proj<<<dim3(BATCH, NPOS / 16, 7), dim3(128), 0, stream>>>(
      x, Wq, bq, Wk, bk, Wv, bv, q, kp, vT);

  const size_t smem = (size_t)(NPOS * 16 + 256 + 16 + 16) * sizeof(float);
  k_energy_softmax<<<dim3(BATCH, NPOS / 16), dim3(256), smem, stream>>>(
      q, kp, Wt);

  k_attn_out<<<dim3(BATCH, NPOS / 32, CH / 32), dim3(32), 0, stream>>>(
      Wt, vT, x, gamma, y);
}
